// MentionScore_88983132439048
// MI455X (gfx1250) — matrix operand order repaired and checked by reading, not compiled, not run on hardware
//
#include <hip/hip_runtime.h>
#include <hip/hip_bf16.h>
#include <stdint.h>

// ---------------------------------------------------------------------------
// Mention scoring, factored form:
//   attns = MLP_a(states)                       (WMMA bf16 GEMMs, fused last layer)
//   P1 = states @ sw1[0:A], P2 = states @ sw1[A:2A], Q = embeds @ sw1[2A:]
//   h1[n] = relu(P1[s_n] + P2[e_n] + sum_w p_w Q[idx_w] + sb1)   (span kernel)
//   out   = relu(h1 @ sw2 + sb2) @ sw3 + sb3    (WMMA GEMM, fused last layer)
//
// GEMM operands are padded (N->160, K->multiple of 32) and the weight (B)
// matrices are stored PRE-TRANSPOSED as Bt[160][Kpad] at conversion time, so
// the WMMA GEMM does only 128-bit global loads + 128-bit LDS stores for both
// A and B staging, aligned ds_load_b128 fragment reads, and 10 WMMAs per
// wave per K-step. No bounds checks anywhere in the hot loop.
// ---------------------------------------------------------------------------

typedef __attribute__((ext_vector_type(16))) __bf16 v16bf;
typedef __attribute__((ext_vector_type(8)))  float  v8f;

#define T_TOK  8192
#define A_DIM  1024
#define E_DIM  512
#define N_SPAN 65536
#define H_DIM  150
#define HP     160        // padded hidden dim
#define MAXW_  10

__device__ __forceinline__ unsigned short f2bf(float f) {
  unsigned u = __float_as_uint(f);
  unsigned r = (u + 0x7FFFu + ((u >> 16) & 1u)) >> 16;   // round-to-nearest-even
  return (unsigned short)r;
}

__global__ void convert_f32_bf16(const float* __restrict__ src,
                                 unsigned short* __restrict__ dst, int n) {
  int i = blockIdx.x * blockDim.x + threadIdx.x;
  int stride = gridDim.x * blockDim.x;
  for (; i < n; i += stride) dst[i] = f2bf(src[i]);
}

// src is [K,150] row-major f32 weights; dst is TRANSPOSED padded bf16
// Bt[160][Kpad]: Bt[n*Kpad + k] = src[k*150 + n], zeros beyond K rows / 150 cols.
__global__ void convert_pad_matT(const float* __restrict__ src,
                                 unsigned short* __restrict__ dst,
                                 int K, int Kpad) {
  int n_elems = HP * Kpad;
  int i = blockIdx.x * blockDim.x + threadIdx.x;
  int stride = gridDim.x * blockDim.x;
  for (; i < n_elems; i += stride) {
    int n = i / Kpad, k = i - n * Kpad;
    dst[i] = (k < K && n < H_DIM) ? f2bf(src[(size_t)k * H_DIM + n])
                                  : (unsigned short)0;
  }
}

// pad a [150] f32 vector to [160] f32 with zeros
__global__ void pad_vec160(const float* __restrict__ src, float* __restrict__ dst) {
  int i = blockIdx.x * blockDim.x + threadIdx.x;
  if (i < HP) dst[i] = (i < H_DIM) ? src[i] : 0.0f;
}

__global__ void fill_scalar(float* __restrict__ dst, const float* __restrict__ s, int n) {
  int i = blockIdx.x * blockDim.x + threadIdx.x;
  int stride = gridDim.x * blockDim.x;
  float v = s[0];
  for (; i < n; i += stride) dst[i] = v;
}

// ---------------------------------------------------------------------------
// WMMA GEMM: C[M,160] = A[M,K] @ B[K,160]  (bf16 in, f32 accumulate)
// Bt is the pre-transposed weight [160][Kpad] (fragment-friendly layout).
// Requirements: M % 128 == 0, K % 32 == 0, no OOB anywhere (padded operands).
// Block: 256 threads = 8 waves as 4(M) x 2(N). Wave tile: 32 x 80 = 10 WMMAs.
// Epilogue: +bias[n], ReLU, store f32/bf16 (ld=160), or fuse final dot with w3
// (padded, zeros beyond 150) and atomicAdd per-row partials into accum[m].
// ---------------------------------------------------------------------------
__global__ __launch_bounds__(256)
void wmma_gemm_bf16(const unsigned short* __restrict__ A, int lda,
                    const unsigned short* __restrict__ Bt, int ldbK,
                    const float* __restrict__ bias,
                    float* __restrict__ Cf, unsigned short* __restrict__ Cb,
                    int M, int K, int relu,
                    const float* __restrict__ w3, float* __restrict__ accum) {
  __shared__ unsigned short Ash[128 * 32];   // A tile [row][k]
  __shared__ unsigned short BshT[HP * 32];   // B tile transposed [n][k]

  const int t    = threadIdx.x;
  const int wave = t >> 5;
  const int lane = t & 31;
  const int half = lane >> 4;      // 0: lanes 0-15, 1: lanes 16-31
  const int r    = lane & 15;
  const int wm   = wave & 3;       // M quarter: rows wm*32 .. +31
  const int wn   = wave >> 2;      // N half: cols wn*80 .. +79

  const int m_block = blockIdx.x * 128;

  v8f acc[2][5] = {};

  // staging coordinates (A): 16 bf16 per thread, two uint4s
  const int srow = t >> 1;
  const int sh   = t & 1;

  for (int k0 = 0; k0 < K; k0 += 32) {
    // ---- stage A: 128x32 bf16, unguarded 128-bit path ----
    {
      const uint4* gp =
          (const uint4*)(A + (size_t)(m_block + srow) * lda + k0 + sh * 16);
      uint4* dst = (uint4*)&Ash[srow * 32 + sh * 16];
      dst[0] = gp[0];
      dst[1] = gp[1];
      if (k0 + 32 < K)   // prefetch next K-step's A rows into near cache
        __builtin_prefetch(A + (size_t)(m_block + srow) * lda + k0 + 32 + sh * 16, 0, 3);
    }
    // ---- stage B^T slab: BshT[n][k0..k0+32), pure 128-bit copies ----
    {
#pragma unroll
      for (int j = 0; j < 3; ++j) {
        int chunk = t + 256 * j;          // 8-bf16 chunks, 640 total
        if (chunk < 640) {
          int n   = chunk >> 2;           // 0..159
          int pos = (chunk & 3) * 8;      // 0,8,16,24
          *(uint4*)&BshT[n * 32 + pos] =
              *(const uint4*)(Bt + (size_t)n * ldbK + k0 + pos);
        }
      }
    }
    __syncthreads();

    // ---- fragments per CDNA5 16-bit WMMA VGPR layouts ----
    union U { unsigned u[8]; v16bf v; };
    U ua[2];
#pragma unroll
    for (int mi = 0; mi < 2; ++mi) {
      int row = (wm * 32 + mi * 16 + r) * 32;
#pragma unroll
      for (int i = 0; i < 8; ++i) {
        int kk = ((i < 4) ? (2 * i) : (16 + 2 * (i - 4))) + half * 8;
        ua[mi].u[i] = *(const unsigned*)&Ash[row + kk];
      }
    }
#pragma unroll
    for (int nt = 0; nt < 5; ++nt) {
      U ub;
      int col = (wn * 80 + nt * 16 + r) * 32;
#pragma unroll
      for (int i = 0; i < 8; ++i) {
        int kk = ((i < 4) ? (2 * i) : (16 + 2 * (i - 4))) + half * 8;
        ub.u[i] = *(const unsigned*)&BshT[col + kk];
      }
      acc[0][nt] = __builtin_amdgcn_wmma_f32_16x16x32_bf16(
          false, ua[0].v, false, ub.v, (short)0, acc[0][nt], false, false);
      acc[1][nt] = __builtin_amdgcn_wmma_f32_16x16x32_bf16(
          false, ua[1].v, false, ub.v, (short)0, acc[1][nt], false, false);
    }
    __syncthreads();
  }

  // ---- epilogue: C/D layout = lane column r, VGPR i -> row i + 8*half ----
#pragma unroll
  for (int mi = 0; mi < 2; ++mi) {
    float pacc[8] = {0, 0, 0, 0, 0, 0, 0, 0};
#pragma unroll
    for (int nt = 0; nt < 5; ++nt) {
      int n = wn * 80 + nt * 16 + r;
#pragma unroll
      for (int i = 0; i < 8; ++i) {
        int m = m_block + wm * 32 + mi * 16 + half * 8 + i;
        float v = acc[mi][nt][i];
        if (bias) v += bias[n];
        if (relu) v = fmaxf(v, 0.0f);
        if (Cf) Cf[(size_t)m * HP + n] = v;
        if (Cb) Cb[(size_t)m * HP + n] = f2bf(v);
        if (w3) pacc[i] += v * w3[n];
      }
    }
    if (w3) {
#pragma unroll
      for (int i = 0; i < 8; ++i) {
        float p = pacc[i];
#pragma unroll
        for (int off = 8; off >= 1; off >>= 1)
          p += __shfl_xor(p, off, 32);        // stays within each 16-lane half
        if (r == 0)
          atomicAdd(&accum[m_block + wm * 32 + mi * 16 + half * 8 + i], p);
      }
    }
  }
}

// ---------------------------------------------------------------------------
// One wave per span: masked softmax over <=10 logits, then
//   h1 = relu(P1[s] + P2[e] + sum_j p_j * Q[s+j] + sb1)  -> bf16 row of G1
// Padding columns [150,160) are written as zero so the following K=160 GEMM
// needs no guards.
// ---------------------------------------------------------------------------
__global__ __launch_bounds__(256)
void span_kernel(const float* __restrict__ attns,
                 const float* __restrict__ P1,
                 const float* __restrict__ P2,
                 const float* __restrict__ Q,
                 const float* __restrict__ sb1,
                 const int* __restrict__ starts,
                 const int* __restrict__ widths,
                 unsigned short* __restrict__ G1) {
  const int wave = threadIdx.x >> 5;
  const int lane = threadIdx.x & 31;
  const int n = blockIdx.x * 8 + wave;
  if (n >= N_SPAN) return;

  const int s = starts[n];
  const int wr = widths[n];
  const int e = s + wr;          // inclusive end
  const int count = wr + 1;      // 1..10 tokens

  float a = (lane < count) ? attns[s + lane] : -3.0e38f;
  float mx = a;
#pragma unroll
  for (int off = 16; off >= 1; off >>= 1)
    mx = fmaxf(mx, __shfl_xor(mx, off, 32));
  float ex = (lane < count) ? __expf(a - mx) : 0.0f;
  float sum = ex;
#pragma unroll
  for (int off = 16; off >= 1; off >>= 1)
    sum += __shfl_xor(sum, off, 32);
  float prob = ex / sum;

  float pj[MAXW_];
#pragma unroll
  for (int j = 0; j < MAXW_; ++j) pj[j] = __shfl(prob, j, 32);

#pragma unroll
  for (int it = 0; it < 5; ++it) {
    int f = lane + it * 32;      // covers 0..159
    float acc = 0.0f;
    if (f < H_DIM) {
      acc = sb1[f] + P1[(size_t)s * HP + f] + P2[(size_t)e * HP + f];
      for (int j = 0; j < count; ++j)
        acc += pj[j] * Q[(size_t)(s + j) * HP + f];
      acc = fmaxf(acc, 0.0f);
    }
    G1[(size_t)n * HP + f] = f2bf(acc);
  }
}

// ---------------------------------------------------------------------------
extern "C" void kernel_launch(void* const* d_in, const int* in_sizes, int n_in,
                              void* d_out, int out_size, void* d_ws, size_t ws_size,
                              hipStream_t stream) {
  (void)in_sizes; (void)n_in; (void)out_size; (void)ws_size;

  const float* states      = (const float*)d_in[0];
  const float* embeds      = (const float*)d_in[1];
  const int*   span_starts = (const int*)d_in[2];
  const int*   span_widths = (const int*)d_in[3];
  const float* aw1 = (const float*)d_in[4];
  const float* ab1 = (const float*)d_in[5];
  const float* aw2 = (const float*)d_in[6];
  const float* ab2 = (const float*)d_in[7];
  const float* aw3 = (const float*)d_in[8];
  const float* ab3 = (const float*)d_in[9];
  const float* sw1 = (const float*)d_in[10];
  const float* sb1 = (const float*)d_in[11];
  const float* sw2 = (const float*)d_in[12];
  const float* sb2 = (const float*)d_in[13];
  const float* sw3 = (const float*)d_in[14];
  const float* sb3 = (const float*)d_in[15];
  float* out = (float*)d_out;

  // --- workspace carve (aligned 256B) ---
  char* ws = (char*)d_ws;
  size_t off = 0;
  auto carve = [&](size_t bytes) -> void* {
    void* p = ws + off;
    off = (off + bytes + 255) & ~(size_t)255;
    return p;
  };
  unsigned short* states_bf = (unsigned short*)carve((size_t)T_TOK * A_DIM * 2);
  unsigned short* embeds_bf = (unsigned short*)carve((size_t)T_TOK * E_DIM * 2);
  unsigned short* aw1_t  = (unsigned short*)carve((size_t)HP * A_DIM * 2);
  unsigned short* aw2_t  = (unsigned short*)carve((size_t)HP * HP * 2);
  unsigned short* sw1a_t = (unsigned short*)carve((size_t)HP * A_DIM * 2);
  unsigned short* sw1b_t = (unsigned short*)carve((size_t)HP * A_DIM * 2);
  unsigned short* sw1c_t = (unsigned short*)carve((size_t)HP * E_DIM * 2);
  unsigned short* sw2_t  = (unsigned short*)carve((size_t)HP * HP * 2);
  float* ab1_pad = (float*)carve(HP * 4);
  float* ab2_pad = (float*)carve(HP * 4);
  float* sb2_pad = (float*)carve(HP * 4);
  float* aw3_pad = (float*)carve(HP * 4);
  float* sw3_pad = (float*)carve(HP * 4);
  unsigned short* h1_bf = (unsigned short*)carve((size_t)T_TOK * HP * 2);
  float* attns = (float*)carve((size_t)T_TOK * 4);
  float* P1 = (float*)carve((size_t)T_TOK * HP * 4);
  float* P2 = (float*)carve((size_t)T_TOK * HP * 4);
  float* Q  = (float*)carve((size_t)T_TOK * HP * 4);
  unsigned short* G1 = (unsigned short*)carve((size_t)N_SPAN * HP * 2);

  // --- conversions (activations plain; weights transposed+padded [160][Kpad]) ---
  convert_f32_bf16<<<512, 256, 0, stream>>>(states, states_bf, T_TOK * A_DIM);
  convert_f32_bf16<<<256, 256, 0, stream>>>(embeds, embeds_bf, T_TOK * E_DIM);
  convert_pad_matT<<<160, 256, 0, stream>>>(aw1, aw1_t, A_DIM, A_DIM);
  convert_pad_matT<<<25,  256, 0, stream>>>(aw2, aw2_t, H_DIM, HP);
  convert_pad_matT<<<160, 256, 0, stream>>>(sw1, sw1a_t, A_DIM, A_DIM);
  convert_pad_matT<<<160, 256, 0, stream>>>(sw1 + (size_t)A_DIM * H_DIM, sw1b_t, A_DIM, A_DIM);
  convert_pad_matT<<<80,  256, 0, stream>>>(sw1 + (size_t)2 * A_DIM * H_DIM, sw1c_t, E_DIM, E_DIM);
  convert_pad_matT<<<25,  256, 0, stream>>>(sw2, sw2_t, H_DIM, HP);
  pad_vec160<<<1, HP, 0, stream>>>(ab1, ab1_pad);
  pad_vec160<<<1, HP, 0, stream>>>(ab2, ab2_pad);
  pad_vec160<<<1, HP, 0, stream>>>(sb2, sb2_pad);
  pad_vec160<<<1, HP, 0, stream>>>(aw3, aw3_pad);
  pad_vec160<<<1, HP, 0, stream>>>(sw3, sw3_pad);

  // --- init accumulators with final-layer biases ---
  fill_scalar<<<32,  256, 0, stream>>>(attns, ab3, T_TOK);
  fill_scalar<<<256, 256, 0, stream>>>(out, sb3, N_SPAN);

  dim3 blk(256);

  // attention layer 1: h1 = relu(states @ aw1 + ab1)   [T,160] bf16
  wmma_gemm_bf16<<<dim3(T_TOK / 128), blk, 0, stream>>>(
      states_bf, A_DIM, aw1_t, A_DIM, ab1_pad, nullptr, h1_bf,
      T_TOK, A_DIM, 1, nullptr, nullptr);
  // attention layers 2+3 fused: attns += relu(h1 @ aw2 + ab2) @ aw3
  wmma_gemm_bf16<<<dim3(T_TOK / 128), blk, 0, stream>>>(
      h1_bf, HP, aw2_t, HP, ab2_pad, nullptr, nullptr,
      T_TOK, HP, 1, aw3_pad, attns);

  // P1 = states @ sw1[0:A]   [T,160] f32
  wmma_gemm_bf16<<<dim3(T_TOK / 128), blk, 0, stream>>>(
      states_bf, A_DIM, sw1a_t, A_DIM, nullptr, P1, nullptr,
      T_TOK, A_DIM, 0, nullptr, nullptr);
  // P2 = states @ sw1[A:2A]
  wmma_gemm_bf16<<<dim3(T_TOK / 128), blk, 0, stream>>>(
      states_bf, A_DIM, sw1b_t, A_DIM, nullptr, P2, nullptr,
      T_TOK, A_DIM, 0, nullptr, nullptr);
  // Q = embeds @ sw1[2A:]
  wmma_gemm_bf16<<<dim3(T_TOK / 128), blk, 0, stream>>>(
      embeds_bf, E_DIM, sw1c_t, E_DIM, nullptr, Q, nullptr,
      T_TOK, E_DIM, 0, nullptr, nullptr);

  // span softmax-pool + assemble mention-MLP input  -> G1 [N,160] bf16
  span_kernel<<<N_SPAN / 8, 256, 0, stream>>>(attns, P1, P2, Q, sb1,
                                              span_starts, span_widths, G1);

  // mention layers 2+3 fused: out += relu(G1 @ sw2 + sb2) @ sw3
  wmma_gemm_bf16<<<dim3(N_SPAN / 128), blk, 0, stream>>>(
      G1, HP, sw2_t, HP, sb2_pad, nullptr, nullptr,
      N_SPAN, HP, 1, sw3_pad, out);
}
